// Cross_AgentAttention_12343736008875
// MI455X (gfx1250) — matrix operand order repaired and checked by reading, not compiled, not run on hardware
//
#include <hip/hip_runtime.h>
#include <hip/hip_bf16.h>

// ---------------------------------------------------------------------------
// Agent attention for MI455X (gfx1250, wave32, WMMA).
// All heavy matmuls use v_wmma_f32_16x16x32_bf16 with f32 accumulation.
// ---------------------------------------------------------------------------

using bf16 = __bf16;
typedef __bf16 v16bf __attribute__((ext_vector_type(16)));
typedef __bf16 v8bf  __attribute__((ext_vector_type(8)));
typedef __bf16 v4bf  __attribute__((ext_vector_type(4)));
typedef float  v8f   __attribute__((ext_vector_type(8)));
typedef float  v4f   __attribute__((ext_vector_type(4)));

#define B_     16
#define N_     4096
#define DIM_   512
#define HEADS_ 8
#define DH_    64
#define INNER_ 512
#define AGENT_ 49
#define APAD_  64
#define HW_    64
#define SCALE_ 0.125f

__device__ __forceinline__ v8f wmma_bf16(v16bf a, v16bf b, v8f c) {
  return __builtin_amdgcn_wmma_f32_16x16x32_bf16(false, a, false, b, (short)0, c,
                                                 false, false);
}

__device__ __forceinline__ v16bf cat16(v8bf lo, v8bf hi) {
  v16bf r;
#pragma unroll
  for (int i = 0; i < 8; ++i) { r[i] = lo[i]; r[8 + i] = hi[i]; }
  return r;
}

__device__ __forceinline__ v8f vzero8() {
  v8f z;
#pragma unroll
  for (int i = 0; i < 8; ++i) z[i] = 0.0f;
  return z;
}

// ---------------------------------------------------------------------------
// Bias precompute: bilinear resize 7x7 -> 64x64 (half-pixel, edge clamp, as
// jax.image.resize 'bilinear') + separable biases.
// ---------------------------------------------------------------------------
__device__ __forceinline__ float bilerp7(const float* __restrict__ g, int yi, int xi) {
  float fy = (yi + 0.5f) * (7.0f / 64.0f) - 0.5f;
  float fx = (xi + 0.5f) * (7.0f / 64.0f) - 0.5f;
  int y0 = (int)floorf(fy); float wy = fy - (float)y0;
  int x0 = (int)floorf(fx); float wx = fx - (float)x0;
  int y0c = min(max(y0, 0), 6), y1c = min(max(y0 + 1, 0), 6);
  int x0c = min(max(x0, 0), 6), x1c = min(max(x0 + 1, 0), 6);
  float a = g[y0c * 7 + x0c], b = g[y0c * 7 + x1c];
  float c = g[y1c * 7 + x0c], d = g[y1c * 7 + x1c];
  return (1.0f - wy) * ((1.0f - wx) * a + wx * b) + wy * ((1.0f - wx) * c + wx * d);
}

// bias1[h][a(64)][n=4096] = resize(an_bias)[h][a][y][x] + ah[h][a][y] + aw[h][a][x]
__global__ __launch_bounds__(256) void bias1_kernel(
    const float* __restrict__ an_bias, const float* __restrict__ ah_bias,
    const float* __restrict__ aw_bias, float* __restrict__ bias1) {
  int idx = blockIdx.x * 256 + threadIdx.x;           // 8*64*4096
  int n = idx & 4095, a = (idx >> 12) & 63, h = idx >> 18;
  float v = 0.0f;
  if (a < AGENT_) {
    int y = n >> 6, x = n & 63;
    v = bilerp7(an_bias + (h * AGENT_ + a) * 49, y, x)
        + ah_bias[(h * AGENT_ + a) * HW_ + y]
        + aw_bias[(h * AGENT_ + a) * HW_ + x];
  }
  bias1[idx] = v;
}

// bias2[h][n=4096][a(64)] = resize(na_bias)[h][a][y][x] + ha[h][y][a] + wa[h][x][a]
__global__ __launch_bounds__(256) void bias2_kernel(
    const float* __restrict__ na_bias, const float* __restrict__ ha_bias,
    const float* __restrict__ wa_bias, float* __restrict__ bias2) {
  int idx = blockIdx.x * 256 + threadIdx.x;           // 8*4096*64
  int a = idx & 63, n = (idx >> 6) & 4095, h = idx >> 18;
  float v = 0.0f;
  if (a < AGENT_) {
    int y = n >> 6, x = n & 63;
    v = bilerp7(na_bias + (h * AGENT_ + a) * 49, y, x)
        + ha_bias[(h * HW_ + y) * AGENT_ + a]
        + wa_bias[(h * HW_ + x) * AGENT_ + a];
  }
  bias2[idx] = v;
}

// ---------------------------------------------------------------------------
// Generic C[M,N] = A[M,K] @ W[N,K]^T + bias, bf16 WMMA core.
// Block tile 128x128x32, 256 threads = 8 waves (4 along M x 2 along N);
// each wave computes a 32x64 patch = 2x4 v_wmma_f32_16x16x32_bf16 tiles.
// Double-buffered LDS, register prefetch, ONE barrier per K step, plus
// global_prefetch two K tiles ahead.
// ---------------------------------------------------------------------------
template <typename T> struct vec4_of;
template <> struct vec4_of<float> { using type = v4f; };
template <> struct vec4_of<bf16>  { using type = v4bf; };

__device__ __forceinline__ void st4(bf16* dst, v4f d) {
#pragma unroll
  for (int j = 0; j < 4; ++j) dst[j] = (bf16)d[j];
}
__device__ __forceinline__ void st4(bf16* dst, v4bf d) {
#pragma unroll
  for (int j = 0; j < 4; ++j) dst[j] = d[j];
}
__device__ __forceinline__ void store_out(float* C, size_t i, float v) { C[i] = v; }
__device__ __forceinline__ void store_out(bf16* C, size_t i, float v) { C[i] = (bf16)v; }

template <typename AT, typename OT>
__global__ __launch_bounds__(256) void gemm_bias_kernel(
    const AT* __restrict__ A, const float* __restrict__ W,
    const float* __restrict__ bias, OT* __restrict__ C,
    int M, int Ncols, int K) {
  constexpr int BM = 128, BN = 128, BK = 32, LDK = 40;  // LDK*2B=80B, 16B multiple
  __shared__ alignas(16) bf16 As[2][BM * LDK];
  __shared__ alignas(16) bf16 Ws[2][BN * LDK];
  using AV = typename vec4_of<AT>::type;

  const int tid = threadIdx.x, lane = tid & 31, wave = tid >> 5;
  const int bm = blockIdx.x * BM, bn = blockIdx.y * BN;
  const int wm = (wave & 3) * 32, wn = (wave >> 2) * 64;

  v8f acc[2][4];
#pragma unroll
  for (int i = 0; i < 2; ++i)
#pragma unroll
    for (int j = 0; j < 4; ++j) acc[i][j] = vzero8();

  const int arow = lane & 15, ak0 = (lane < 16) ? 0 : 16;  // A: K {0-15}|{16-31}
  const int bcol = lane & 15, bk0 = (lane < 16) ? 0 : 16;  // B: K {0-15}|{16-31}

  // per-thread staging coords: 128 rows x 8 vec4 columns, 4 rows per thread
  const int r0 = tid >> 3, c0 = (tid & 7) * 4;
  const AT*    Aptr = A + (size_t)(bm + r0) * K + c0;
  const float* Wptr = W + (size_t)(bn + r0) * K + c0;

  AV  ra[4];
  v4f rw[4];
#pragma unroll
  for (int i = 0; i < 4; ++i) ra[i] = *(const AV*)(Aptr + (size_t)(32 * i) * K);
#pragma unroll
  for (int i = 0; i < 4; ++i) rw[i] = *(const v4f*)(Wptr + (size_t)(32 * i) * K);
#pragma unroll
  for (int i = 0; i < 4; ++i) {
    st4(&As[0][(r0 + 32 * i) * LDK + c0], ra[i]);
    st4(&Ws[0][(r0 + 32 * i) * LDK + c0], rw[i]);
  }
  __syncthreads();

  const int nk = K / BK;
  for (int kt = 0; kt < nk; ++kt) {
    if (kt + 1 < nk) {
      const int ko = (kt + 1) * BK;
#pragma unroll
      for (int i = 0; i < 4; ++i) ra[i] = *(const AV*)(Aptr + (size_t)(32 * i) * K + ko);
#pragma unroll
      for (int i = 0; i < 4; ++i) rw[i] = *(const v4f*)(Wptr + (size_t)(32 * i) * K + ko);
      if (kt + 2 < nk) {   // pull the tile after next toward the caches
        __builtin_prefetch((const void*)(Aptr + (size_t)(kt + 2) * BK), 0, 1);
        __builtin_prefetch((const void*)(Wptr + (size_t)(kt + 2) * BK), 0, 1);
      }
    }

    const bf16* as = As[kt & 1];
    const bf16* ws = Ws[kt & 1];
    v16bf aop[2], bop[4];
#pragma unroll
    for (int t = 0; t < 2; ++t) {
      const bf16* pa = &as[(wm + t * 16 + arow) * LDK + ak0];
      aop[t] = cat16(*(const v8bf*)pa, *(const v8bf*)(pa + 8));
    }
#pragma unroll
    for (int t = 0; t < 4; ++t) {
      const bf16* pb = &ws[(wn + t * 16 + bcol) * LDK + bk0];
      bop[t] = cat16(*(const v8bf*)pb, *(const v8bf*)(pb + 8));
    }
#pragma unroll
    for (int tm = 0; tm < 2; ++tm)
#pragma unroll
      for (int tn = 0; tn < 4; ++tn)
        acc[tm][tn] = wmma_bf16(aop[tm], bop[tn], acc[tm][tn]);

    if (kt + 1 < nk) {
      bf16* asn = As[(kt + 1) & 1];
      bf16* wsn = Ws[(kt + 1) & 1];
#pragma unroll
      for (int i = 0; i < 4; ++i) {
        st4(&asn[(r0 + 32 * i) * LDK + c0], ra[i]);
        st4(&wsn[(r0 + 32 * i) * LDK + c0], rw[i]);
      }
      __syncthreads();
    }
  }

  const int rb = (lane < 16) ? 0 : 8;
#pragma unroll
  for (int tm = 0; tm < 2; ++tm)
#pragma unroll
    for (int tn = 0; tn < 4; ++tn) {
      int n = bn + wn + tn * 16 + (lane & 15);
      float bv = bias[n];
#pragma unroll
      for (int r = 0; r < 8; ++r) {
        size_t m = (size_t)(bm + wm + tm * 16 + rb + r);
        store_out(C, m * (size_t)Ncols + n, acc[tm][tn][r] + bv);
      }
    }
}

// ---------------------------------------------------------------------------
// Adaptive avg-pool q[b, 64x64, C] -> agent[b, 64(pad), C]; pads zeroed.
// ---------------------------------------------------------------------------
__global__ __launch_bounds__(256) void pool_kernel(const bf16* __restrict__ q,
                                                   bf16* __restrict__ agent) {
  int idx = blockIdx.x * 256 + threadIdx.x;          // 16*64*512
  int c = idx & 511, a = (idx >> 9) & 63, b = idx >> 15;
  float v = 0.0f;
  if (a < AGENT_) {
    int ay = a / 7, ax = a % 7;
    int sy = (ay * HW_) / 7, ey = ((ay + 1) * HW_ + 6) / 7;
    int sx = (ax * HW_) / 7, ex = ((ax + 1) * HW_ + 6) / 7;
    float s = 0.0f;
    for (int y = sy; y < ey; ++y)
      for (int x = sx; x < ex; ++x)
        s += (float)q[((size_t)b * N_ + y * HW_ + x) * INNER_ + c];
    v = s / (float)((ey - sy) * (ex - sx));
  }
  agent[idx] = (bf16)v;
}

// ---------------------------------------------------------------------------
// vT[b][h][d][n] = kv[b][n][INNER + h*64 + d]  (tiled LDS transpose)
// ---------------------------------------------------------------------------
__global__ __launch_bounds__(256) void vtrans_kernel(const bf16* __restrict__ kv,
                                                     bf16* __restrict__ vT) {
  int blk = blockIdx.x;                                // 16*8*64
  int nt = blk & 63, h = (blk >> 6) & 7, b = blk >> 9;
  __shared__ bf16 t[64][72];
  int tid = threadIdx.x;
#pragma unroll
  for (int p = 0; p < 16; ++p) {
    int e = p * 256 + tid; int r = e >> 6, c = e & 63;
    t[r][c] = kv[((size_t)b * N_ + nt * 64 + r) * (2 * INNER_) + INNER_ + h * DH_ + c];
  }
  __syncthreads();
#pragma unroll
  for (int p = 0; p < 16; ++p) {
    int e = p * 256 + tid; int r = e >> 6, c = e & 63;
    vT[(((size_t)b * HEADS_ + h) * DH_ + r) * N_ + nt * 64 + c] = t[c][r];
  }
}

// ---------------------------------------------------------------------------
// Stage 1: agents attend to k/v (flash-softmax over n2).
// One block per (b,h): 8 waves = 4 row groups x 2 n2 halves; halves merged in LDS.
// Output pre-transposed: agent_vT[b][h][d][a].
// ---------------------------------------------------------------------------
__global__ __launch_bounds__(256) void stage1_kernel(
    const bf16* __restrict__ agent, const bf16* __restrict__ kv,
    const bf16* __restrict__ vT, const float* __restrict__ bias1,
    bf16* __restrict__ agent_vT) {
  const int h = blockIdx.x & 7, b = blockIdx.x >> 3;
  const int tid = threadIdx.x, lane = tid & 31, wave = tid >> 5;
  const int rowbase = (wave & 3) * 16;
  const int half = wave >> 2;
  __shared__ alignas(16) bf16 Pbuf[8][16 * 72];
  __shared__ float accs[4][16][68];
  __shared__ float mlbuf[4][2][16];

  // agent A operands (fixed for all chunks): rows rowbase..rowbase+15, K=d (64)
  const bf16* abase = agent + ((size_t)b * APAD_ + rowbase + (lane & 15)) * INNER_ + h * DH_;
  const int ak0 = (lane < 16) ? 0 : 16;
  v16bf a0 = cat16(*(const v8bf*)(abase + ak0), *(const v8bf*)(abase + ak0 + 8));
  v16bf a1 = cat16(*(const v8bf*)(abase + 32 + ak0), *(const v8bf*)(abase + 32 + ak0 + 8));

  v8f acc[4];
#pragma unroll
  for (int i = 0; i < 4; ++i) acc[i] = vzero8();
  float mrow[8], lrow[8];
#pragma unroll
  for (int r = 0; r < 8; ++r) { mrow[r] = -3.0e38f; lrow[r] = 0.0f; }

  const int bcol = lane & 15, bk = (lane < 16) ? 0 : 16;
  const int rb = (lane < 16) ? 0 : 8;
  bf16* pb = Pbuf[wave];

  for (int jc = 0; jc < 32; ++jc) {
    const int j0 = half * 2048 + jc * 64;
    v8f S[4];
#pragma unroll
    for (int t = 0; t < 4; ++t) {          // S = (agent @ k^T) * SCALE + bias1
      const bf16* kp = kv + ((size_t)b * N_ + j0 + t * 16 + bcol) * (2 * INNER_) + h * DH_;
      v16bf b0 = *(const v16bf*)(kp + bk);
      v16bf b1 = *(const v16bf*)(kp + 32 + bk);
      v8f c = vzero8();
      c = wmma_bf16(a0, b0, c);
      c = wmma_bf16(a1, b1, c);
      const float* bp = bias1 + ((size_t)h * APAD_ + rowbase) * N_ + (j0 + t * 16 + bcol);
#pragma unroll
      for (int r = 0; r < 8; ++r) c[r] = c[r] * SCALE_ + bp[(size_t)(rb + r) * N_];
      S[t] = c;
    }
    // row-wise max across 4 tiles + 16-lane halves
    float alpha[8], rs[8];
#pragma unroll
    for (int r = 0; r < 8; ++r) {
      float v = fmaxf(fmaxf(S[0][r], S[1][r]), fmaxf(S[2][r], S[3][r]));
#pragma unroll
      for (int msk = 1; msk < 16; msk <<= 1) v = fmaxf(v, __shfl_xor(v, msk));
      float mnew = fmaxf(mrow[r], v);
      alpha[r] = __expf(mrow[r] - mnew);
      mrow[r] = mnew;
      rs[r] = 0.0f;
    }
    // P = exp(S - m) -> LDS (C-layout scatter), accumulate row sums
#pragma unroll
    for (int t = 0; t < 4; ++t)
#pragma unroll
      for (int r = 0; r < 8; ++r) {
        float p = __expf(S[t][r] - mrow[r]);
        rs[r] += p;
        pb[(r + rb) * 72 + t * 16 + bcol] = (bf16)p;
      }
#pragma unroll
    for (int r = 0; r < 8; ++r) {
#pragma unroll
      for (int msk = 1; msk < 16; msk <<= 1) rs[r] += __shfl_xor(rs[r], msk);
      lrow[r] = lrow[r] * alpha[r] + rs[r];
#pragma unroll
      for (int dt = 0; dt < 4; ++dt) acc[dt][r] *= alpha[r];
    }
    __syncthreads();   // uniform trip count across all 8 waves

    // read P back in WMMA A-layout, multiply by V (vT rows contiguous in n)
    const bf16* prow = pb + (lane & 15) * 72;
    v16bf p0 = cat16(*(const v8bf*)(prow + ak0), *(const v8bf*)(prow + ak0 + 8));
    v16bf p1 = cat16(*(const v8bf*)(prow + 32 + ak0), *(const v8bf*)(prow + 32 + ak0 + 8));
#pragma unroll
    for (int dt = 0; dt < 4; ++dt) {
      const bf16* vp = vT + (((size_t)b * HEADS_ + h) * DH_ + dt * 16 + bcol) * N_ + j0 + bk;
      v16bf vb0 = *(const v16bf*)vp;
      v16bf vb1 = *(const v16bf*)(vp + 32);
      acc[dt] = wmma_bf16(p0, vb0, acc[dt]);
      acc[dt] = wmma_bf16(p1, vb1, acc[dt]);
    }
  }

  // merge the two n2-halves
  if (wave >= 4) {
    int w = wave - 4;
#pragma unroll
    for (int dt = 0; dt < 4; ++dt)
#pragma unroll
      for (int r = 0; r < 8; ++r) accs[w][r + rb][dt * 16 + bcol] = acc[dt][r];
    if ((lane & 15) == 0) {
#pragma unroll
      for (int r = 0; r < 8; ++r) {
        mlbuf[w][0][r + rb] = mrow[r];
        mlbuf[w][1][r + rb] = lrow[r];
      }
    }
  }
  __syncthreads();
  if (wave < 4) {
#pragma unroll
    for (int r = 0; r < 8; ++r) {
      float m2 = mlbuf[wave][0][r + rb];
      float l2 = mlbuf[wave][1][r + rb];
      float M = fmaxf(mrow[r], m2);
      float f1 = __expf(mrow[r] - M), f2 = __expf(m2 - M);
      float inv = 1.0f / (lrow[r] * f1 + l2 * f2);
#pragma unroll
      for (int dt = 0; dt < 4; ++dt) {
        float v = (acc[dt][r] * f1 + accs[wave][r + rb][dt * 16 + bcol] * f2) * inv;
        agent_vT[(((size_t)b * HEADS_ + h) * DH_ + dt * 16 + bcol) * APAD_ +
                 rowbase + r + rb] = (bf16)v;
      }
    }
  }
}

// ---------------------------------------------------------------------------
// Stage 2: queries attend to agents. Block = (b, h, 128 q-rows); wave = 16 rows.
// Single-chunk softmax over 64 padded agents (cols >= 49 masked to -inf).
// ---------------------------------------------------------------------------
__global__ __launch_bounds__(256) void stage2_kernel(
    const bf16* __restrict__ q, const bf16* __restrict__ agent,
    const bf16* __restrict__ agent_vT, const float* __restrict__ bias2,
    bf16* __restrict__ attn_out) {
  const int blk = blockIdx.x;                  // 16*8*32
  const int nb = blk & 31, h = (blk >> 5) & 7, b = blk >> 8;
  const int tid = threadIdx.x, lane = tid & 31, wave = tid >> 5;
  const int row0 = nb * 128 + wave * 16;
  __shared__ alignas(16) bf16 Pbuf[8][16 * 72];

  const bf16* qb = q + ((size_t)b * N_ + row0 + (lane & 15)) * INNER_ + h * DH_;
  const int ak0 = (lane < 16) ? 0 : 16;
  v16bf a0 = cat16(*(const v8bf*)(qb + ak0), *(const v8bf*)(qb + ak0 + 8));
  v16bf a1 = cat16(*(const v8bf*)(qb + 32 + ak0), *(const v8bf*)(qb + 32 + ak0 + 8));

  const int bcol = lane & 15, bk = (lane < 16) ? 0 : 16;
  const int rb = (lane < 16) ? 0 : 8;

  v8f S[4];
#pragma unroll
  for (int t = 0; t < 4; ++t) {                // S2 = (q @ agent^T)*SCALE + bias2
    const bf16* ap = agent + ((size_t)b * APAD_ + t * 16 + bcol) * INNER_ + h * DH_;
    v16bf b0 = *(const v16bf*)(ap + bk);
    v16bf b1 = *(const v16bf*)(ap + 32 + bk);
    v8f c = vzero8();
    c = wmma_bf16(a0, b0, c);
    c = wmma_bf16(a1, b1, c);
    int a_col = t * 16 + bcol;
    const float* bp = bias2 + ((size_t)h * N_ + row0) * APAD_ + a_col;
#pragma unroll
    for (int r = 0; r < 8; ++r) {
      float v = c[r] * SCALE_ + bp[(size_t)(r + rb) * APAD_];
      c[r] = (a_col < AGENT_) ? v : -3.0e38f;
    }
    S[t] = c;
  }

  float rs[8];
  bf16* pb = Pbuf[wave];
#pragma unroll
  for (int r = 0; r < 8; ++r) {
    float mx = fmaxf(fmaxf(S[0][r], S[1][r]), fmaxf(S[2][r], S[3][r]));
#pragma unroll
    for (int msk = 1; msk < 16; msk <<= 1) mx = fmaxf(mx, __shfl_xor(mx, msk));
    float s = 0.0f;
#pragma unroll
    for (int t = 0; t < 4; ++t) {
      float p = __expf(S[t][r] - mx);          // masked cols -> exp(-huge)=0
      s += p;
      pb[(r + rb) * 72 + t * 16 + bcol] = (bf16)p;
    }
#pragma unroll
    for (int msk = 1; msk < 16; msk <<= 1) s += __shfl_xor(s, msk);
    rs[r] = s;
  }
  __syncthreads();

  const bf16* prow = pb + (lane & 15) * 72;
  v16bf p0 = cat16(*(const v8bf*)(prow + ak0), *(const v8bf*)(prow + ak0 + 8));
  v16bf p1 = cat16(*(const v8bf*)(prow + 32 + ak0), *(const v8bf*)(prow + 32 + ak0 + 8));

  v8f o[4];
#pragma unroll
  for (int dt = 0; dt < 4; ++dt) {
    const bf16* vp = agent_vT + (((size_t)b * HEADS_ + h) * DH_ + dt * 16 + bcol) * APAD_ + bk;
    v16bf v0 = *(const v16bf*)vp;
    v16bf v1 = *(const v16bf*)(vp + 32);
    o[dt] = vzero8();
    o[dt] = wmma_bf16(p0, v0, o[dt]);
    o[dt] = wmma_bf16(p1, v1, o[dt]);
  }
#pragma unroll
  for (int dt = 0; dt < 4; ++dt)
#pragma unroll
    for (int r = 0; r < 8; ++r) {
      float v = o[dt][r] / rs[r];
      attn_out[((size_t)b * N_ + row0 + r + rb) * INNER_ + h * DH_ + dt * 16 + bcol] =
          (bf16)v;
    }
}

// ---------------------------------------------------------------------------
extern "C" void kernel_launch(void* const* d_in, const int* in_sizes, int n_in,
                              void* d_out, int out_size, void* d_ws, size_t ws_size,
                              hipStream_t stream) {
  const float* x       = (const float*)d_in[0];
  const float* m       = (const float*)d_in[1];
  const float* Wq      = (const float*)d_in[2];
  const float* bq      = (const float*)d_in[3];
  const float* Wkv     = (const float*)d_in[4];
  const float* bkv     = (const float*)d_in[5];
  const float* Wp      = (const float*)d_in[6];
  const float* bp      = (const float*)d_in[7];
  const float* an_bias = (const float*)d_in[8];
  const float* na_bias = (const float*)d_in[9];
  const float* ah_bias = (const float*)d_in[10];
  const float* aw_bias = (const float*)d_in[11];
  const float* ha_bias = (const float*)d_in[12];
  const float* wa_bias = (const float*)d_in[13];
  float* out = (float*)d_out;

  char* w = (char*)d_ws;
  bf16* q        = (bf16*)w;  w += (size_t)B_ * N_ * INNER_ * 2;            // 64 MB
  bf16* kv       = (bf16*)w;  w += (size_t)B_ * N_ * 2 * INNER_ * 2;        // 128 MB
  bf16* vT       = (bf16*)w;  w += (size_t)B_ * HEADS_ * DH_ * N_ * 2;      // 64 MB
  bf16* agent    = (bf16*)w;  w += (size_t)B_ * APAD_ * INNER_ * 2;         // 1 MB
  bf16* agent_vT = (bf16*)w;  w += (size_t)B_ * HEADS_ * DH_ * APAD_ * 2;   // 1 MB
  float* bias1   = (float*)w; w += (size_t)HEADS_ * APAD_ * N_ * 4;         // 8 MB
  float* bias2   = (float*)w; w += (size_t)HEADS_ * N_ * APAD_ * 4;         // 8 MB
  bf16* attn     = (bf16*)w;  w += (size_t)B_ * N_ * INNER_ * 2;            // 64 MB

  const int M = B_ * N_;  // 65536

  bias1_kernel<<<(HEADS_ * APAD_ * N_) / 256, 256, 0, stream>>>(an_bias, ah_bias,
                                                                aw_bias, bias1);
  bias2_kernel<<<(HEADS_ * N_ * APAD_) / 256, 256, 0, stream>>>(na_bias, ha_bias,
                                                                wa_bias, bias2);

  dim3 gq(M / 128, INNER_ / 128);
  gemm_bias_kernel<float, bf16><<<gq, 256, 0, stream>>>(x, Wq, bq, q, M, INNER_, DIM_);
  dim3 gkv(M / 128, (2 * INNER_) / 128);
  gemm_bias_kernel<float, bf16><<<gkv, 256, 0, stream>>>(m, Wkv, bkv, kv, M,
                                                         2 * INNER_, DIM_);

  pool_kernel<<<(B_ * APAD_ * INNER_) / 256, 256, 0, stream>>>(q, agent);
  vtrans_kernel<<<B_ * HEADS_ * (N_ / 64), 256, 0, stream>>>(kv, vT);

  stage1_kernel<<<B_ * HEADS_, 256, 0, stream>>>(agent, kv, vT, bias1, agent_vT);
  stage2_kernel<<<B_ * HEADS_ * (N_ / 128), 256, 0, stream>>>(q, agent, agent_vT,
                                                              bias2, attn);

  dim3 go(M / 128, DIM_ / 128);
  gemm_bias_kernel<bf16, float><<<go, 256, 0, stream>>>(attn, Wp, bp, out, M, DIM_,
                                                        DIM_);
}
